// MyModel_20667382628498
// MI455X (gfx1250) — matrix operand order repaired and checked by reading, not compiled, not run on hardware
//
#include <hip/hip_runtime.h>
#include <math.h>

// ---------------- Types for CDNA5 WMMA / TDM ----------------
typedef __bf16 bf16_t;
typedef bf16_t v16bf __attribute__((ext_vector_type(16)));
typedef float  v8f   __attribute__((ext_vector_type(8)));
typedef unsigned int u32;
typedef u32    u32x4 __attribute__((ext_vector_type(4)));
typedef int    i32x8 __attribute__((ext_vector_type(8)));
typedef int    i32x4 __attribute__((ext_vector_type(4)));

struct LayerP {
    const bf16_t* Wt;   // [Cout][CinPad] bf16, K-contiguous per column
    const float*  b;
    const float*  g;
    const float*  be;
    const float*  rm;
    const float*  rv;
};

__device__ __forceinline__ v16bf ldfrag(const bf16_t* p, int o0, int o1) {
    union { v16bf v; u32x4 q[2]; } f;
    f.q[0] = *(const u32x4*)(p + o0);
    f.q[1] = *(const u32x4*)(p + o1);
    return f.v;
}

__device__ __forceinline__ v8f wmma_bf16(v16bf a, v16bf b, v8f c) {
    return __builtin_amdgcn_wmma_f32_16x16x32_bf16(false, a, false, b, (short)0, c,
                                                   false, false);
}

// One 16x16 C tile: A row fragment (LDS), B column fragment (global bf16)
template<int CINP>
__device__ __forceinline__ v8f tile_mm(const bf16_t* arow, const bf16_t* wcol, int h) {
    v8f acc = {0.f, 0.f, 0.f, 0.f, 0.f, 0.f, 0.f, 0.f};
#pragma unroll
    for (int kt = 0; kt < CINP / 32; ++kt) {
        v16bf a = ldfrag(arow + kt * 32, 8 * h, 16 + 8 * h);          // 16-bit A layout
        v16bf b = ldfrag(wcol + kt * 32, 8 * h, 16 + 8 * h);          // 16-bit B layout
        acc = wmma_bf16(a, b, acc);
    }
    return acc;
}

__device__ __forceinline__ float bn_relu(float v, float bb, float gg, float be,
                                         float rm, float inv) {
    v = fmaxf(v + bb, 0.f);
    return gg * (v - rm) * inv + be;
}

// LDS-resident layer: 64 rows x CINP -> 64 rows x COUT, one m-tile per wave
template<int CINP, int COUT>
__device__ __forceinline__ void layer_lds(const bf16_t* in, const LayerP& L,
                                          bf16_t* out, int wave, int lane) {
    const int h  = lane >> 4;
    const int nl = lane & 15;
    const bf16_t* arow = in + (wave * 16 + nl) * CINP;
    for (int nt = 0; nt < COUT / 16; ++nt) {
        const int col = nt * 16 + nl;
        v8f acc = tile_mm<CINP>(arow, L.Wt + col * CINP, h);
        const float bb = L.b[col], gg = L.g[col], be = L.be[col], rm = L.rm[col];
        const float inv = rsqrtf(L.rv[col] + 1e-5f);
#pragma unroll
        for (int r = 0; r < 8; ++r) {
            int row = wave * 16 + r + 8 * h;
            out[row * COUT + col] = (bf16_t)bn_relu(acc[r], bb, gg, be, rm, inv);
        }
    }
}

// ---------------- Tensor Data Mover helpers (gfx1250 D# descriptors) ----------------
// This toolchain's builtin is the 6-arg form: (g0, g1, g2, g3, g4, cpol).
__device__ __forceinline__ void tdm_issue(u32x4 g0, i32x8 g1, i32x4 g2, i32x4 g3) {
    i32x8 g4 = {0, 0, 0, 0, 0, 0, 0, 0};
    __builtin_amdgcn_tensor_load_to_lds(g0, g1, g2, g3, g4, 0);
}

// Normal-mode 2D tile load: rows x CINP bf16 elements, row stride = CINP.
__device__ __forceinline__ void tdm_load_tile_2d(unsigned ldsAddr,
                                                 unsigned long long gAddr,
                                                 int cinp, int rows) {
    u32x4 g0; i32x8 g1; i32x4 g2 = {0, 0, 0, 0}, g3 = {0, 0, 0, 0};
    g0[0] = 1u;                                           // count=1, load descriptor
    g0[1] = ldsAddr;                                      // LDS byte address
    g0[2] = (u32)(gAddr & 0xffffffffull);                 // global addr [31:0]
    g0[3] = (u32)((gAddr >> 32) & 0x1ffffffull) | (2u << 30);  // addr[56:32] | type=2
    g1[0] = (int)(1u << 16);                              // data_size = 2 bytes (bf16)
    g1[1] = (int)((u32)cinp << 16);                       // tensor_dim0[15:0]
    g1[2] = (int)((u32)rows << 16);                       // tensor_dim1[15:0]
    g1[3] = (int)((u32)cinp << 16);                       // tile_dim0
    g1[4] = rows;                                         // tile_dim1
    g1[5] = cinp;                                         // tensor_dim0_stride[31:0]
    g1[6] = 0; g1[7] = 0;
    tdm_issue(g0, g1, g2, g3);
}

// Gather-mode load: 16 rows (16-bit indices) of a [Np x XCH] bf16 tensor into LDS,
// padding each row from XCH to CINP channels via pad_interval/pad_amount.
template<int XCH, int CINP>
__device__ __forceinline__ void tdm_gather16(unsigned ldsAddr,
                                             unsigned long long gBase,
                                             int Np, const int* idx16) {
    // payload per row: XCH bf16 = XCH/2 dwords; pad: (CINP-XCH)/2 dwords
    constexpr unsigned padDw = (unsigned)(CINP - XCH) / 2;       // dwords of pad
    constexpr unsigned intDw = (unsigned)XCH / 2;                // dwords between pads
    constexpr unsigned padIntervalCode = __builtin_ctz(intDw) - 1;  // 2^(c+1)=intDw
    constexpr unsigned padAmountCode   = padDw - 1;
    u32x4 g0; i32x8 g1; i32x4 g2, g3;
    g0[0] = 0x80000001u;                                  // count=1 | gather_mode(31)
    g0[1] = ldsAddr;
    g0[2] = (u32)(gBase & 0xffffffffull);
    g0[3] = (u32)((gBase >> 32) & 0x1ffffffull) | (2u << 30);
    g1[0] = (int)((1u << 16) | (1u << 20) |               // data_size=2B | pad_enable
                  (padIntervalCode << 22) | (padAmountCode << 25));
    g1[1] = (int)((u32)XCH << 16);                        // tensor_dim0 = XCH
    g1[2] = (int)((u32)Np << 16);                         // tensor_dim1 = Np
    g1[3] = (int)((u32)XCH << 16);                        // tile_dim0 = XCH
    g1[4] = 16;                                           // tile_dim1 = #valid indices
    g1[5] = XCH;                                          // tensor_dim0_stride
    g1[6] = 0; g1[7] = 0;
    g2[0] = idx16[0]; g2[1] = idx16[1]; g2[2] = idx16[2]; g2[3] = idx16[3];
    g3[0] = idx16[4]; g3[1] = idx16[5]; g3[2] = idx16[6]; g3[3] = idx16[7];
    tdm_issue(g0, g1, g2, g3);
}

// ---------------- Weight conversion: fp32 [Cin,Cout] -> bf16 [Cout][CinPad] ----------------
__global__ void conv_w(const float* W, bf16_t* Wt, int Cin, int Cout, int CinPad) {
    int i = blockIdx.x * blockDim.x + threadIdx.x;
    if (i >= Cout * CinPad) return;
    int col = i / CinPad, k = i - col * CinPad;
    Wt[i] = (bf16_t)((k < Cin) ? W[k * Cout + col] : 0.f);
}

// ---------------- Farthest point sampling (1 block per graph) ----------------
__global__ __launch_bounds__(256) void fps_kernel(const float* pos, int Np, int m,
                                                  int* idxOut, float* centOut) {
    __shared__ float dmin[2048];
    __shared__ float rv_[256];
    __shared__ int   ri_[256];
    __shared__ float lx, ly;
    const int b = blockIdx.x, tid = threadIdx.x;
    const float* P = pos + (size_t)b * Np * 2;
    const float x0 = P[0], y0 = P[1];
    for (int i = tid; i < Np; i += 256) {
        float dx = P[i * 2] - x0, dy = P[i * 2 + 1] - y0;
        dmin[i] = dx * dx + dy * dy;
    }
    if (tid == 0) {
        idxOut[b * m] = 0;
        centOut[(size_t)b * m * 2] = x0;
        centOut[(size_t)b * m * 2 + 1] = y0;
        lx = x0; ly = y0;
    }
    __syncthreads();
    for (int it = 1; it < m; ++it) {
        float bv = -1.f; int bi = 0x7fffffff;
        const float cxl = lx, cyl = ly;
        for (int i = tid; i < Np; i += 256) {
            float dx = P[i * 2] - cxl, dy = P[i * 2 + 1] - cyl;
            float d = fminf(dmin[i], dx * dx + dy * dy);
            dmin[i] = d;
            if (d > bv || (d == bv && i < bi)) { bv = d; bi = i; }
        }
        rv_[tid] = bv; ri_[tid] = bi;
        __syncthreads();
        for (int s = 128; s > 0; s >>= 1) {
            if (tid < s) {
                float v2 = rv_[tid + s]; int i2 = ri_[tid + s];
                if (v2 > rv_[tid] || (v2 == rv_[tid] && i2 < ri_[tid])) {
                    rv_[tid] = v2; ri_[tid] = i2;
                }
            }
            __syncthreads();
        }
        if (tid == 0) {
            int bidx = ri_[0];
            idxOut[b * m + it] = bidx;
            float nx = P[bidx * 2], ny = P[bidx * 2 + 1];
            centOut[((size_t)b * m + it) * 2] = nx;
            centOut[((size_t)b * m + it) * 2 + 1] = ny;
            lx = nx; ly = ny;
        }
        __syncthreads();
    }
}

// ---------------- radius-kNN: top-64 nearest within r (1 block per center) ----------------
__global__ __launch_bounds__(256) void knn_kernel(const float* pos, const float* cent,
                                                  int Np, int m, float r2,
                                                  int* nbr, int* mask) {
    __shared__ float d2s[2048];
    __shared__ float rv_[256];
    __shared__ int   ri_[256];
    const int blk = blockIdx.x, tid = threadIdx.x;
    const int bg = blk / m;
    const float* P = pos + (size_t)bg * Np * 2;
    const float cx = cent[(size_t)blk * 2], cy = cent[(size_t)blk * 2 + 1];
    for (int i = tid; i < Np; i += 256) {
        float dx = P[i * 2] - cx, dy = P[i * 2 + 1] - cy;
        float d = dx * dx + dy * dy;
        d2s[i] = (d <= r2) ? d : __builtin_inff();
    }
    __syncthreads();
    for (int k = 0; k < 64; ++k) {
        float bv = __builtin_inff(); int bi = 0x7fffffff;
        for (int i = tid; i < Np; i += 256) {
            float d = d2s[i];
            if (d < bv || (d == bv && i < bi)) { bv = d; bi = i; }
        }
        rv_[tid] = bv; ri_[tid] = bi;
        __syncthreads();
        for (int s = 128; s > 0; s >>= 1) {
            if (tid < s) {
                float v2 = rv_[tid + s]; int i2 = ri_[tid + s];
                if (v2 < rv_[tid] || (v2 == rv_[tid] && i2 < ri_[tid])) {
                    rv_[tid] = v2; ri_[tid] = i2;
                }
            }
            __syncthreads();
        }
        if (tid == 0) {
            if (rv_[0] < __builtin_inff()) {
                nbr[(size_t)blk * 64 + k] = ri_[0];
                mask[(size_t)blk * 64 + k] = 1;
                d2s[ri_[0]] = __builtin_inff();
            } else {
                nbr[(size_t)blk * 64 + k] = 0;
                mask[(size_t)blk * 64 + k] = 0;
            }
        }
        __syncthreads();
    }
}

// ---------------- Fused SA module: gather -> 3-layer WMMA MLP -> masked max ----------------
template<int CINP, int C1, int C2, int C3, int XCH, bool XF32>
__global__ __launch_bounds__(128) void sa_fused(
    const float* xf, const bf16_t* xb,
    const float* pos, int Np,
    const float* centers, const int* nbr, const int* mask, int m,
    LayerP L1, LayerP L2, LayerP L3, bf16_t* out)
{
    constexpr int E_F = 64 * CINP, E_H0 = 64 * C1, E_H1 = 64 * C2, E_H2 = 64 * C3;
    constexpr bool LINEAR = (E_F + E_H0 + E_H1 + E_H2) <= 32768;   // 64KB of bf16
    constexpr int OFF_F  = 0;
    constexpr int OFF_H0 = E_F;
    constexpr int OFF_H1 = LINEAR ? (E_F + E_H0) : 0;              // reuse feats region
    constexpr int OFF_H2 = LINEAR ? (E_F + E_H0 + E_H1) : E_H1;    // reuse h0 region
    constexpr int P_A = E_F + E_H0;
    constexpr int P_B = E_H1 + E_H2;
    constexpr int POOL = LINEAR ? (E_F + E_H0 + E_H1 + E_H2) : (P_A > P_B ? P_A : P_B);

    __shared__ __align__(16) bf16_t pool[POOL];
    __shared__ int msk_s[64];
    __shared__ int nbr_s[64];

    bf16_t* feats = pool + OFF_F;
    bf16_t* h0    = pool + OFF_H0;
    bf16_t* h1    = pool + OFF_H1;
    bf16_t* h2    = pool + OFF_H2;

    const int blk = blockIdx.x, tid = threadIdx.x;
    const int bg = blk / m;
    const float cx = centers[(size_t)blk * 2], cy = centers[(size_t)blk * 2 + 1];

    if (tid < 64) {
        msk_s[tid] = mask[(size_t)blk * 64 + tid];
        nbr_s[tid] = nbr[(size_t)blk * 64 + tid];
    }
    __syncthreads();

    if constexpr (!XF32) {
        // TDM gather-mode: 4 descriptors x 16 row-indices pull the 64 neighbor
        // feature rows ([Np x XCH] bf16) into LDS, padded to CINP channels.
        if (tid < 32) {
            const unsigned ldsBase = (unsigned)(uintptr_t)feats;
            const unsigned long long gBase =
                (unsigned long long)(uintptr_t)xb +
                (unsigned long long)bg * (unsigned)Np * (unsigned)XCH * sizeof(bf16_t);
            for (int gsec = 0; gsec < 4; ++gsec) {
                int p[8];
#pragma unroll
                for (int t = 0; t < 8; ++t) {
                    int i0 = nbr_s[gsec * 16 + 2 * t];
                    int i1 = nbr_s[gsec * 16 + 2 * t + 1];
                    p[t] = __builtin_amdgcn_readfirstlane((i0 & 0xffff) | (i1 << 16));
                }
                tdm_gather16<XCH, CINP>(
                    ldsBase + (unsigned)gsec * 16u * CINP * sizeof(bf16_t),
                    gBase, Np, p);
            }
            __builtin_amdgcn_s_wait_tensorcnt(0);
        }
        __syncthreads();
        // Fill relative-position channels + zero the pad region (TDM skipped it)
        constexpr int PADW = CINP - XCH;
        for (int i = tid; i < 64 * PADW; i += 128) {
            int e = i / PADW, ch = XCH + (i - e * PADW);
            float v = 0.f;
            int j = nbr_s[e];
            if (ch == XCH)          v = pos[((size_t)bg * Np + j) * 2] - cx;
            else if (ch == XCH + 1) v = pos[((size_t)bg * Np + j) * 2 + 1] - cy;
            feats[e * CINP + ch] = (bf16_t)v;
        }
    } else {
        // fp32 source (SA1): VALU gather [x_j | p_j - center | 0-pad] as bf16
        for (int i = tid; i < 64 * CINP; i += 128) {
            int e = i / CINP, ch = i - e * CINP;
            float v = 0.f;
            if (msk_s[e]) {
                int j = nbr_s[e];
                if (ch < XCH)           v = xf[((size_t)bg * Np + j) * XCH + ch];
                else if (ch == XCH)     v = pos[((size_t)bg * Np + j) * 2] - cx;
                else if (ch == XCH + 1) v = pos[((size_t)bg * Np + j) * 2 + 1] - cy;
            }
            feats[e * CINP + ch] = (bf16_t)v;
        }
    }
    __syncthreads();

    const int wave = tid >> 5, lane = tid & 31;
    layer_lds<CINP, C1>(feats, L1, h0, wave, lane);
    __syncthreads();
    layer_lds<C1, C2>(h0, L2, h1, wave, lane);
    __syncthreads();
    layer_lds<C2, C3>(h1, L3, h2, wave, lane);
    __syncthreads();

    // Masked max over 64 edges (center itself is always in-radius -> >=1 valid)
    for (int ch = tid; ch < C3; ch += 128) {
        float best = -__builtin_inff();
        for (int e = 0; e < 64; ++e)
            if (msk_s[e]) best = fmaxf(best, (float)h2[e * C3 + ch]);
        out[(size_t)blk * C3 + ch] = (bf16_t)best;
    }
}

// ---------------- Global-module row GEMM: out = BN(ReLU(in @ W + b)) ----------------
// A tile (16 x CINP) staged once into LDS via normal-mode TDM, reused by all waves.
template<int CINP, int COUT>
__global__ __launch_bounds__(128) void gemm_bn(const bf16_t* in, LayerP L, bf16_t* out) {
    __shared__ __align__(16) bf16_t atile[16 * CINP];
    const int rt = blockIdx.x;
    if (threadIdx.x < 32) {
        tdm_load_tile_2d((unsigned)(uintptr_t)atile,
                         (unsigned long long)(uintptr_t)in +
                             (unsigned long long)rt * 16u * CINP * sizeof(bf16_t),
                         CINP, 16);
        __builtin_amdgcn_s_wait_tensorcnt(0);
    }
    __syncthreads();
    const int wave = threadIdx.x >> 5, lane = threadIdx.x & 31;
    const int h = lane >> 4, nl = lane & 15;
    const bf16_t* arow = atile + nl * CINP;
    for (int nt = wave; nt < COUT / 16; nt += 4) {
        const int col = nt * 16 + nl;
        v8f acc = tile_mm<CINP>(arow, L.Wt + col * CINP, h);
        const float bb = L.b[col], gg = L.g[col], be = L.be[col], rm = L.rm[col];
        const float inv = rsqrtf(L.rv[col] + 1e-5f);
#pragma unroll
        for (int r = 0; r < 8; ++r) {
            int row = rt * 16 + r + 8 * h;
            out[(size_t)row * COUT + col] = (bf16_t)bn_relu(acc[r], bb, gg, be, rm, inv);
        }
    }
}

// ---------------- Concat [x2 | p2 | pad] -> feats3 ----------------
__global__ void concat3(const bf16_t* x2, const float* cent2, bf16_t* f3) {
    int i = blockIdx.x * blockDim.x + threadIdx.x;
    if (i >= 2048 * 288) return;
    int r = i / 288, ch = i - r * 288;
    float v = 0.f;
    if (ch < 256)      v = (float)x2[(size_t)r * 256 + ch];
    else if (ch < 258) v = cent2[(size_t)r * 2 + (ch - 256)];
    f3[i] = (bf16_t)v;
}

// ---------------- Global max pool over 256 rows per graph ----------------
__global__ void maxpool_k(const bf16_t* h, float* g) {
    int i = blockIdx.x * blockDim.x + threadIdx.x;
    if (i >= 8 * 1024) return;
    int b = i >> 10, ch = i & 1023;
    float best = -__builtin_inff();
    for (int r = 0; r < 256; ++r)
        best = fmaxf(best, (float)h[((size_t)(b << 8) + r) * 1024 + ch]);
    g[i] = best;
}

// ---------------- FC head (tiny, fp32): 1024->512->256->1, sigmoid ----------------
__global__ __launch_bounds__(256) void fc_head(const float* g,
                                               const float* W1, const float* b1,
                                               const float* W2, const float* b2,
                                               const float* W3, const float* b3,
                                               float* out) {
    __shared__ float s1[512];
    __shared__ float s2[256];
    __shared__ float red[256];
    const int b = blockIdx.x, tid = threadIdx.x;
    const float* gv = g + (size_t)b * 1024;
    for (int c = tid; c < 512; c += 256) {
        float acc = b1[c];
        for (int k = 0; k < 1024; ++k) acc += gv[k] * W1[(size_t)k * 512 + c];
        s1[c] = fmaxf(acc, 0.f);
    }
    __syncthreads();
    for (int c = tid; c < 256; c += 256) {
        float acc = b2[c];
        for (int k = 0; k < 512; ++k) acc += s1[k] * W2[(size_t)k * 256 + c];
        s2[c] = fmaxf(acc, 0.f);
    }
    __syncthreads();
    red[tid] = s2[tid] * W3[tid];
    __syncthreads();
    for (int s = 128; s > 0; s >>= 1) {
        if (tid < s) red[tid] += red[tid + s];
        __syncthreads();
    }
    if (tid == 0) out[b] = 1.f / (1.f + expf(-(red[0] + b3[0])));
}

// ---------------- Launcher ----------------
extern "C" void kernel_launch(void* const* d_in, const int* in_sizes, int n_in,
                              void* d_out, int out_size, void* d_ws, size_t ws_size,
                              hipStream_t stream) {
    (void)in_sizes; (void)n_in; (void)out_size; (void)ws_size;
    auto F = [&](int i) { return (const float*)d_in[i]; };
    const float* x   = F(0);   // [8*2048, 3]
    const float* pos = F(1);   // [8*2048, 2]

    char* w = (char*)d_ws;
    auto alloc = [&](size_t bytes) -> char* {
        char* p = w;
        w += (bytes + 255) & ~(size_t)255;
        return p;
    };

    // Convert all 9 MLP weight matrices to bf16 [Cout][CinPad]
    static const int cin_[9]  = {5, 64, 64, 130, 128, 128, 258, 256, 512};
    static const int cout_[9] = {64, 64, 128, 128, 128, 256, 256, 512, 1024};
    static const int cinp_[9] = {32, 64, 64, 160, 128, 128, 288, 256, 512};
    static const int widx_[9] = {3, 9, 15, 21, 27, 33, 39, 45, 51};
    bf16_t* Wt[9];
    for (int i = 0; i < 9; ++i) {
        int n = cout_[i] * cinp_[i];
        Wt[i] = (bf16_t*)alloc((size_t)n * sizeof(bf16_t));
        conv_w<<<(n + 255) / 256, 256, 0, stream>>>(F(widx_[i]), Wt[i],
                                                    cin_[i], cout_[i], cinp_[i]);
    }

    int*    idx1  = (int*)alloc((size_t)8 * 1024 * 4);
    float*  cent1 = (float*)alloc((size_t)8 * 1024 * 2 * 4);
    int*    nbr1  = (int*)alloc((size_t)8 * 1024 * 64 * 4);
    int*    msk1  = (int*)alloc((size_t)8 * 1024 * 64 * 4);
    bf16_t* x1    = (bf16_t*)alloc((size_t)8 * 1024 * 128 * 2);
    int*    idx2  = (int*)alloc((size_t)8 * 256 * 4);
    float*  cent2 = (float*)alloc((size_t)8 * 256 * 2 * 4);
    int*    nbr2  = (int*)alloc((size_t)8 * 256 * 64 * 4);
    int*    msk2  = (int*)alloc((size_t)8 * 256 * 64 * 4);
    bf16_t* x2    = (bf16_t*)alloc((size_t)8 * 256 * 256 * 2);
    bf16_t* f3    = (bf16_t*)alloc((size_t)2048 * 288 * 2);
    bf16_t* h31   = (bf16_t*)alloc((size_t)2048 * 256 * 2);
    bf16_t* h32   = (bf16_t*)alloc((size_t)2048 * 512 * 2);
    bf16_t* h33   = (bf16_t*)alloc((size_t)2048 * 1024 * 2);
    float*  gp    = (float*)alloc((size_t)8 * 1024 * 4);

    auto LP = [&](int i) -> LayerP {
        int base = widx_[i];
        return LayerP{Wt[i], F(base + 1), F(base + 2), F(base + 3), F(base + 4), F(base + 5)};
    };

    // ---- SA module 1: ratio 0.5, r 0.2 ----
    fps_kernel<<<8, 256, 0, stream>>>(pos, 2048, 1024, idx1, cent1);
    knn_kernel<<<8 * 1024, 256, 0, stream>>>(pos, cent1, 2048, 1024, 0.04f, nbr1, msk1);
    sa_fused<32, 64, 64, 128, 3, true><<<8 * 1024, 128, 0, stream>>>(
        x, nullptr, pos, 2048, cent1, nbr1, msk1, 1024, LP(0), LP(1), LP(2), x1);

    // ---- SA module 2: ratio 0.25, r 0.4 ----
    fps_kernel<<<8, 256, 0, stream>>>(cent1, 1024, 256, idx2, cent2);
    knn_kernel<<<8 * 256, 256, 0, stream>>>(cent1, cent2, 1024, 256, 0.16f, nbr2, msk2);
    sa_fused<160, 128, 128, 256, 128, false><<<8 * 256, 128, 0, stream>>>(
        nullptr, x1, cent1, 1024, cent2, nbr2, msk2, 256, LP(3), LP(4), LP(5), x2);

    // ---- Global SA module ----
    concat3<<<(2048 * 288 + 255) / 256, 256, 0, stream>>>(x2, cent2, f3);
    gemm_bn<288, 256><<<128, 128, 0, stream>>>(f3, LP(6), h31);
    gemm_bn<256, 512><<<128, 128, 0, stream>>>(h31, LP(7), h32);
    gemm_bn<512, 1024><<<128, 128, 0, stream>>>(h32, LP(8), h33);
    maxpool_k<<<32, 256, 0, stream>>>(h33, gp);

    // ---- FC head ----
    fc_head<<<8, 256, 0, stream>>>(gp, F(57), F(58), F(59), F(60), F(61), F(62),
                                   (float*)d_out);
}
